// CovarianceSimilarity_65592740545256
// MI455X (gfx1250) — matrix operand order repaired and checked by reading, not compiled, not run on hardware
//
#include <hip/hip_runtime.h>

// ---- CDNA5 (gfx1250) types for WMMA ----
typedef __bf16 v16bf __attribute__((ext_vector_type(16)));
typedef __bf16 v8bf  __attribute__((ext_vector_type(8)));
typedef float  v8f   __attribute__((ext_vector_type(8)));

#define C_DIM   1024   // channels (GEMM K)
#define D_DIM   196    // spatial H*W
#define D_PAD   208    // 13 * 16 (zero padded)
#define NQ      256
#define SHOT    5
#define KC      512    // K chunk staged in LDS
#define BSTRIDE (KC + 8)  // LDS row stride in bf16 elems: +16B pad -> 4-bank rotate

// -------------------------------------------------------------------------
// Kernel 1: X = center_spatial(mean_shot(support)); store X^T as bf16 hi/lo
// A[a][c] = X[c][a], a in [0,208) (rows >=196 are zero), row-major stride 1024.
// -------------------------------------------------------------------------
__global__ __launch_bounds__(256) void cov_prep(const float* __restrict__ sup,
                                                __bf16* __restrict__ Ahi,
                                                __bf16* __restrict__ Alo) {
    __shared__ float buf[256];
    __shared__ float red[256];
    const int c = blockIdx.x;
    const int t = threadIdx.x;

    float v = 0.0f;
    if (t < D_DIM) {
        float s = 0.0f;
#pragma unroll
        for (int sh = 0; sh < SHOT; ++sh)
            s += sup[sh * (C_DIM * D_DIM) + c * D_DIM + t];
        v = s * (1.0f / (float)SHOT);
    }
    buf[t] = v;
    red[t] = v;
    __syncthreads();
#pragma unroll
    for (int off = 128; off > 0; off >>= 1) {
        if (t < off) red[t] += red[t + off];
        __syncthreads();
    }
    const float mean = red[0] * (1.0f / (float)D_DIM);

    if (t < D_PAD) {
        float x = (t < D_DIM) ? (buf[t] - mean) : 0.0f;
        __bf16 h = (__bf16)x;                 // RNE high part
        __bf16 l = (__bf16)(x - (float)h);    // residual low part
        Ahi[t * C_DIM + c] = h;
        Alo[t * C_DIM + c] = l;
    }
}

// -------------------------------------------------------------------------
// Kernel 2: per (query, n-split) block: partial = sum over 13 m-tiles of
// || (X^T Q_i)[m-tile, n-tile] ||^2 , K = 1024 reduced via bf16x3 WMMA.
// 416 threads = 13 wave32s; wave w owns m-tile w (rows 16w..16w+15).
// -------------------------------------------------------------------------
__global__ __launch_bounds__(416) void cov_gemm(const float* __restrict__ Q,
                                                const __bf16* __restrict__ Ahi,
                                                const __bf16* __restrict__ Alo,
                                                float* __restrict__ partial) {
    __shared__ __bf16 bhi_s[16 * BSTRIDE];
    __shared__ __bf16 blo_s[16 * BSTRIDE];
    __shared__ float  red[512];

    const int ns  = blockIdx.x;   // n-split 0..12  -> columns n0..n0+15 of Q_i
    const int q   = blockIdx.y;   // query 0..255
    const int tid = threadIdx.x;
    const int n0  = ns * 16;
    const float* qbase = Q + (size_t)q * (C_DIM * D_DIM);

    const int wave = tid >> 5;        // 0..12  == m-tile index
    const int lane = tid & 31;
    const int half = lane >> 4;       // lane half selects K sub-block
    const int arow = wave * 16 + (lane & 15);  // A row (M), < 208
    const int nrow = lane & 15;                // B column (N) within tile

    // Fragment base pointers following the CDNA5 16-bit WMMA VGPR layouts:
    //  A 16x32: lanes 0-15 take K {0..7,16..23}, lanes 16-31 take {8..15,24..31}
    //  B 32x16: lanes 0-15 take K {0..7,16..23}, lanes 16-31 take {8..15,24..31}
    const __bf16* ah = Ahi + arow * C_DIM + half * 8;   // +16B for upper half-wave
    const __bf16* al = Alo + arow * C_DIM + half * 8;
    const __bf16* bh = &bhi_s[nrow * BSTRIDE + half * 8]; // +16B for upper half
    const __bf16* bl = &blo_s[nrow * BSTRIDE + half * 8];

    v8f acc = {};

    for (int kc = 0; kc < C_DIM; kc += KC) {
        // ---- stage B chunk: Bt[n][k] = Q_i[kc+k][n0+n] as bf16 hi/lo ----
        for (int idx = tid; idx < 16 * KC; idx += 416) {
            int cl = idx >> 4;        // local k
            int n  = idx & 15;
            float v = (n0 + n < D_DIM) ? qbase[(kc + cl) * D_DIM + n0 + n] : 0.0f;
            __bf16 h = (__bf16)v;
            __bf16 l = (__bf16)(v - (float)h);
            bhi_s[n * BSTRIDE + cl] = h;
            blo_s[n * BSTRIDE + cl] = l;
        }
        __syncthreads();

        // ---- WMMA K loop over this chunk ----
        for (int k = 0; k < KC; k += 32) {
            int kg = kc + k;
            v8bf a0 = *(const v8bf*)(ah + kg);
            v8bf a1 = *(const v8bf*)(ah + kg + 16);
            v16bf aHi = __builtin_shufflevector(a0, a1,
                0,1,2,3,4,5,6,7,8,9,10,11,12,13,14,15);
            a0 = *(const v8bf*)(al + kg);
            a1 = *(const v8bf*)(al + kg + 16);
            v16bf aLo = __builtin_shufflevector(a0, a1,
                0,1,2,3,4,5,6,7,8,9,10,11,12,13,14,15);

            v8bf b0 = *(const v8bf*)(bh + k);
            v8bf b1 = *(const v8bf*)(bh + k + 16);
            v16bf bHi = __builtin_shufflevector(b0, b1,
                0,1,2,3,4,5,6,7,8,9,10,11,12,13,14,15);
            b0 = *(const v8bf*)(bl + k);
            b1 = *(const v8bf*)(bl + k + 16);
            v16bf bLo = __builtin_shufflevector(b0, b1,
                0,1,2,3,4,5,6,7,8,9,10,11,12,13,14,15);

            // bf16x3: hi*hi + hi*lo + lo*hi  (lo*lo term ~2^-16, dropped)
            acc = __builtin_amdgcn_wmma_f32_16x16x32_bf16(
                      false, aHi, false, bHi, (short)0, acc, false, false);
            acc = __builtin_amdgcn_wmma_f32_16x16x32_bf16(
                      false, aHi, false, bLo, (short)0, acc, false, false);
            acc = __builtin_amdgcn_wmma_f32_16x16x32_bf16(
                      false, aLo, false, bHi, (short)0, acc, false, false);
        }
        __syncthreads();   // protect LDS before next chunk restage
    }

    // ---- sum of squares of the Z tile, deterministic block reduction ----
    float ss = 0.0f;
#pragma unroll
    for (int j = 0; j < 8; ++j) ss += acc[j] * acc[j];
    red[tid] = ss;
    if (tid < 96) red[416 + tid] = 0.0f;   // pad 416 -> 512
    __syncthreads();
#pragma unroll
    for (int off = 256; off > 0; off >>= 1) {
        if (tid < off) red[tid] += red[tid + off];
        __syncthreads();
    }
    if (tid == 0) partial[q * 13 + ns] = red[0];
}

// -------------------------------------------------------------------------
// Kernel 3: combine 13 partials per query, apply scale/((d-1+eps)*d)
// -------------------------------------------------------------------------
__global__ __launch_bounds__(256) void cov_final(const float* __restrict__ partial,
                                                 const float* __restrict__ scale,
                                                 float* __restrict__ out) {
    const int q = threadIdx.x;   // 256 queries
    float s = 0.0f;
#pragma unroll
    for (int j = 0; j < 13; ++j) s += partial[q * 13 + j];
    const float factor = scale[0] / ((195.0f + 1e-8f) * 196.0f);
    out[q] = s * factor;
}

// -------------------------------------------------------------------------
extern "C" void kernel_launch(void* const* d_in, const int* in_sizes, int n_in,
                              void* d_out, int out_size, void* d_ws, size_t ws_size,
                              hipStream_t stream) {
    const float* qf    = (const float*)d_in[0];  // (256,1024,14,14) f32
    const float* sup   = (const float*)d_in[1];  // (5,1024,14,14)   f32
    const float* scale = (const float*)d_in[2];  // scalar f32

    // workspace layout (total ~866 KB):
    //   [0)            X^T hi  : 208*1024 bf16 = 425984 B
    //   [425984)       X^T lo  : 425984 B
    //   [851968)       partial : 256*13 f32 = 13312 B
    char* ws = (char*)d_ws;
    __bf16* Ahi    = (__bf16*)(ws);
    __bf16* Alo    = (__bf16*)(ws + 425984);
    float*  part   = (float*)(ws + 851968);
    float*  out    = (float*)d_out;

    cov_prep <<<dim3(1024),    dim3(256), 0, stream>>>(sup, Ahi, Alo);
    cov_gemm <<<dim3(13, 256), dim3(416), 0, stream>>>(qf, Ahi, Alo, part);
    cov_final<<<dim3(1),       dim3(256), 0, stream>>>(part, scale, out);
}
